// LutLayer_3143916061159
// MI455X (gfx1250) — matrix operand order repaired and checked by reading, not compiled, not run on hardware
//
#include <hip/hip_runtime.h>
#include <math.h>

#define BATCH 2048
#define DEPTH 2048
#define SIXC  6
#define ND    16              // d-values per TDM tile
#define NT    8               // tiles per wave (128 d per wave)
#define RS    100             // LDS row stride in dwords (16B aligned, conflict-free)
#define WAVE_BUF (16 * RS)    // dwords per wave buffer (16 b-rows)

typedef _Float16     h2    __attribute__((ext_vector_type(2)));
typedef _Float16     v16h  __attribute__((ext_vector_type(16)));
typedef float        v8f   __attribute__((ext_vector_type(8)));
typedef float        f2v   __attribute__((ext_vector_type(2)));
typedef unsigned int u32x4 __attribute__((ext_vector_type(4)));
typedef int          i32x4 __attribute__((ext_vector_type(4)));
typedef int          i32x8 __attribute__((ext_vector_type(8)));

union AV { v16h v; h2 h[8]; };

// Pass 1: gate[d,i] = sigmoid(50*lut[d,i]) stored as f16 in natural (d*64+i) order.
// This order IS the WMMA B-operand order for our K==i enumeration.
__global__ void lut_gate_kernel(const float* __restrict__ lut,
                                _Float16* __restrict__ gate, int n) {
  int i = blockIdx.x * blockDim.x + threadIdx.x;
  if (i < n) {
    float v = 50.0f * lut[i];
    gate[i] = (_Float16)(1.0f / (1.0f + __expf(-v)));
  }
}

// Pass 2: main kernel. Each wave: 16 b-rows x 128 d, TDM double-buffered tiles of 16 d.
__global__ __launch_bounds__(128) void lut_main_kernel(
    const float* __restrict__ in,       // (BATCH, DEPTH, 6) f32
    const _Float16* __restrict__ gate,  // (DEPTH, 64) f16
    float* __restrict__ out) {          // (BATCH, DEPTH) f32
  __shared__ float ldsx[4 * 2 * WAVE_BUF];   // 4 waves x 2 buffers

  const int tid  = threadIdx.x;
  const int lane = tid & 31;
  const int w    = tid >> 5;
  const int wid  = blockIdx.x * 4 + w;       // 2048 waves total
  const int b0   = (wid >> 4) * 16;          // 128 b-tiles
  const int dg   = wid & 15;                 // 16 d-groups of 128
  const int brow = lane & 15;                // A-matrix row handled by this lane pair
  const int hh   = lane >> 4;                // lane half: supplies s_2 (K bit 3 of A)

  // Escape the LDS array so TDM-written data can't be undef-folded; memory clobber
  // keeps LDS reads ordered against the waits below.
  asm volatile("" : : "v"(&ldsx[0]) : "memory");

  // ---- TDM descriptor (constant part). 2D tensor (2048 rows x 12288 cols of f32),
  // iterate mode: 16 iterations of a 96x1 row-tile, LDS stride 100 dwords/row.
  i32x8 g1;
  g1[0] = 0x000A0000;        // data_size=4B (2<<16), iterate_enable (1<<19)
  g1[1] = 0x30000000;        // tensor_dim0 = 12288 (low16 in [31:16])
  g1[2] = 0x08000000;        // tensor_dim0 hi16=0 | tensor_dim1=2048 low16
  g1[3] = 0x00600000;        // tensor_dim1 hi16=0 | tile_dim0 = 96
  g1[4] = 1;                 // tile_dim1 = 1 (one row per iteration), tile_dim2=0
  g1[5] = DEPTH * SIXC;      // tensor_dim0_stride = 12288 (low32)
  g1[6] = 0;                 // stride0 hi16 = 0, stride1 low16 = 0
  g1[7] = 0;
  i32x4 g2;
  g2[0] = 1;                 // tensor_dim2 = 1 (benign)
  g2[1] = RS;                // lds_addr_increment = 100 elements (400B) per row
  g2[2] = DEPTH * SIXC;      // global_addr_increment = one tensor row
  g2[3] = (15 << 16);        // iterate_count = 15 -> 16 iterations
  i32x4 g3; g3[0] = 0; g3[1] = 0; g3[2] = 0; g3[3] = 0;
  i32x8 g4; g4[0]=0; g4[1]=0; g4[2]=0; g4[3]=0; g4[4]=0; g4[5]=0; g4[6]=0; g4[7]=0;

  auto issue_tdm = [&](int tile, int p) {
    const int d0 = dg * (ND * NT) + tile * ND;
    unsigned long long va =
        (unsigned long long)(in + ((size_t)b0 * DEPTH + d0) * SIXC);
    u32x4 g0;
    g0[0] = 1u;                                        // count=1, user descriptor
    g0[1] = (unsigned)((w * 2 + p) * WAVE_BUF * 4);    // lds_addr (bytes)
    g0[2] = (unsigned)va;                              // global_addr[31:0]
    g0[3] = ((unsigned)(va >> 32) & 0x01FFFFFFu) | 0x80000000u; // addr[56:32]|type=2
    __builtin_amdgcn_tensor_load_to_lds(g0, g1, g2, g3, g4, 0);
  };

  const float eps = 1e-7f;
  issue_tdm(0, 0);

  for (int tile = 0; tile < NT; ++tile) {
    const int p = tile & 1;
    if (tile + 1 < NT) {
      // buffer p^1 was last read in the previous iteration; ensure DS reads drained
      asm volatile("s_wait_dscnt 0" ::: "memory");
      issue_tdm(tile + 1, p ^ 1);
      __builtin_amdgcn_s_wait_tensorcnt(1);   // current tile complete (in-order)
    } else {
      __builtin_amdgcn_s_wait_tensorcnt(0);
    }
    asm volatile("" ::: "memory");            // compiler barrier before LDS reads

    const int d0 = dg * (ND * NT) + tile * ND;
    const int base = (w * 2 + p) * WAVE_BUF + brow * RS;   // even dword index

    for (int dloc = 0; dloc < ND; ++dloc) {
      const int d = d0 + dloc;

      // 6 inputs of element (b0+brow, d) via 3x ds_load_b64 (addrspace(3) path);
      // lanes m and m+16 read the same address (LDS broadcast).
      const f2v* xp = (const f2v*)&ldsx[base + dloc * 6];
      f2v p01 = xp[0];
      f2v p23 = xp[1];
      f2v p45 = xp[2];
      float x0 = p01[0], x1 = p01[1];
      float x2 = p23[0], x3 = p23[1];
      float x4 = p45[0], x5 = p45[1];

      // factors: a_j = relu(x)+eps (bit=0), c_j = relu(1-x)+eps (bit=1)
      float a0 = fmaxf(x0, 0.f) + eps, c0 = fmaxf(1.f - x0, 0.f) + eps;
      float a1 = fmaxf(x1, 0.f) + eps, c1 = fmaxf(1.f - x1, 0.f) + eps;
      float a3 = fmaxf(x3, 0.f) + eps, c3 = fmaxf(1.f - x3, 0.f) + eps;
      float a4 = fmaxf(x4, 0.f) + eps, c4 = fmaxf(1.f - x4, 0.f) + eps;
      float a5 = fmaxf(x5, 0.f) + eps, c5 = fmaxf(1.f - x5, 0.f) + eps;
      float t2 = hh ? (1.f - x2) : x2;        // lane half fixes s_2
      float f2 = fmaxf(t2, 0.f) + eps;

      // product tree over (s0=k5, s1=k4, s3=k2, s4=k1); s5=k0 done packed below
      float u[2];  u[0] = f2 * a0;  u[1] = f2 * c0;
      float vv[4];
#pragma unroll
      for (int t = 0; t < 2; ++t) { vv[t*2+0] = u[t]*a1; vv[t*2+1] = u[t]*c1; }
      float ww[8];
#pragma unroll
      for (int q = 0; q < 4; ++q) { ww[q*2+0] = vv[q]*a3; ww[q*2+1] = vv[q]*c3; }
      float zz[16];
#pragma unroll
      for (int q = 0; q < 8; ++q) { zz[q*2+0] = ww[q]*a4; zz[q*2+1] = ww[q]*c4; }

      // last level in packed f16: dword (t,v) = {z*a5, z*c5}  (k0 = f16 half)
      h2 ac5; ac5[0] = (_Float16)a5; ac5[1] = (_Float16)c5;
      AV A0, A1;
#pragma unroll
      for (int v = 0; v < 8; ++v) {
        h2 z0; z0[0] = (_Float16)zz[v];     z0[1] = z0[0];
        h2 z1; z1[0] = (_Float16)zz[8 + v]; z1[1] = z1[0];
        A0.h[v] = z0 * ac5;                 // v_pk_mul_f16
        A1.h[v] = z1 * ac5;
      }

      // B operand: gate[d, k] in table k order (k = 8h+s, 16+8h+s-8)
      typedef _Float16 v8h __attribute__((ext_vector_type(8)));
      const _Float16* gp = gate + (size_t)d * 64 + hh * 8;
      v16h B0 = __builtin_shufflevector(*(const v8h*)(gp), *(const v8h*)(gp + 16), 0,1,2,3,4,5,6,7,8,9,10,11,12,13,14,15);
      v16h B1 = __builtin_shufflevector(*(const v8h*)(gp + 32), *(const v8h*)(gp + 48), 0,1,2,3,4,5,6,7,8,9,10,11,12,13,14,15);

      v8f cz = {};
      v8f c = __builtin_amdgcn_wmma_f32_16x16x32_f16(
          false, A0.v, false, B0, (short)0, cz, false, false);
      c = __builtin_amdgcn_wmma_f32_16x16x32_f16(
          false, A1.v, false, B1, (short)0, c, false, false);

      // all 16 columns of C are identical = out[b0+M, d]; pick row r=lane&7 (+8*hh)
      const int sel = lane & 7;
      float r0 = (sel & 1) ? c[1] : c[0];
      float r1 = (sel & 1) ? c[3] : c[2];
      float r2 = (sel & 1) ? c[5] : c[4];
      float r3 = (sel & 1) ? c[7] : c[6];
      float s0 = (sel & 2) ? r1 : r0;
      float s1 = (sel & 2) ? r3 : r2;
      float val = (sel & 4) ? s1 : s0;
      if ((lane & 8) == 0) {
        out[((size_t)(b0 + sel + 8 * hh)) * DEPTH + d] = val;
      }
    }
  }
}

extern "C" void kernel_launch(void* const* d_in, const int* in_sizes, int n_in,
                              void* d_out, int out_size, void* d_ws, size_t ws_size,
                              hipStream_t stream) {
  (void)in_sizes; (void)n_in; (void)out_size; (void)ws_size;
  const float* inputs = (const float*)d_in[0];   // (2048, 2048, 6) f32
  const float* lut    = (const float*)d_in[1];   // (2048, 64) f32
  // d_in[2] (p_q_2_lut_table) is a fixed 0/1 selector; its structure is hardcoded.
  float*     out  = (float*)d_out;
  _Float16*  gate = (_Float16*)d_ws;             // 2048*64 f16 = 256KB scratch

  const int ng = DEPTH * 64;
  hipLaunchKernelGGL(lut_gate_kernel, dim3((ng + 255) / 256), dim3(256), 0, stream,
                     lut, gate, ng);
  // 2048 waves = 128 b-tiles x 16 d-groups; 4 waves per block
  hipLaunchKernelGGL(lut_main_kernel, dim3(512), dim3(128), 0, stream,
                     inputs, gate, out);
}
